// HyperbolicImplicationScorer_566935683333
// MI455X (gfx1250) — compile-verified, matrix operand order checked by reading, not run on hardware
//
#include <hip/hip_runtime.h>

// ---------------------------------------------------------------------------
// Types for CDNA5 WMMA (wave32): v_wmma_f32_16x16x32_bf16
//   (bool neg_a, v16bf A, bool neg_b, v16bf B, short c_mod, v8f C, bool, bool)
// ---------------------------------------------------------------------------
typedef __attribute__((ext_vector_type(16))) __bf16 v16bf;
typedef __attribute__((ext_vector_type(8)))  float  v8f;

union Frag16 {
    v16bf v;
    uint4 q[2];
};

// Load a 16-lane x 16-element bf16 fragment for the 16x16x32 WMMA.
// Row-major source, row stride = 256 elements (512 B). Per the ISA 16-bit
// A/B layout: lanes 0-15 hold K = kb+0..7 (regs 0-3) and kb+16..23 (regs 4-7),
// lanes 16-31 hold K = kb+8..15 and kb+24..31. Two 16-byte loads per lane.
__device__ __forceinline__ v16bf load_frag(const unsigned short* __restrict__ rowptr,
                                           int kb, int half) {
    Frag16 f;
    f.q[0] = *reinterpret_cast<const uint4*>(rowptr + kb + half * 8);
    f.q[1] = *reinterpret_cast<const uint4*>(rowptr + kb + 16 + half * 8);
    return f.v;
}

// ---------------------------------------------------------------------------
// Kernel 1: Poincare projection + squared norm + bf16 hi/lo split.
// One 256-thread block per row (D = 256).
//   pi   = v / (1 + sqrt(1 + ||v||^2))
//   n_pi = ||v||^2 * scale^2
//   hi   = bf16_rn(pi), lo = bf16_rn(pi - hi)
// ---------------------------------------------------------------------------
__global__ __launch_bounds__(256) void proj_split_kernel(
    const float* __restrict__ in,
    unsigned short* __restrict__ hi,
    unsigned short* __restrict__ lo,
    float* __restrict__ norm)
{
    __shared__ float red[256];
    const int row = blockIdx.x;
    const int c   = threadIdx.x;

    const float v = in[row * 256 + c];
    red[c] = v * v;
    __syncthreads();
    #pragma unroll
    for (int s = 128; s > 0; s >>= 1) {
        if (c < s) red[c] += red[c + s];
        __syncthreads();
    }
    const float ss    = red[0];
    const float scale = 1.0f / (1.0f + sqrtf(1.0f + ss));
    const float pi    = v * scale;

    // Round-to-nearest bf16 split via bit ops (no __bf16 arithmetic needed).
    const unsigned u  = __float_as_uint(pi);
    const unsigned hb = (u + 0x8000u) & 0xFFFF0000u;
    const float    hf = __uint_as_float(hb);
    const float    lf = pi - hf;
    const unsigned lb = __float_as_uint(lf) + 0x8000u;

    hi[row * 256 + c] = (unsigned short)(hb >> 16);
    lo[row * 256 + c] = (unsigned short)(lb >> 16);
    if (c == 0) norm[row] = ss * scale * scale;
}

// ---------------------------------------------------------------------------
// Kernel 2: split-bf16 WMMA GEMM  dot = pi_x . pi_y  (~f32 accuracy via
// hi*hi + hi*lo + lo*hi) fused with the hyperbolic-distance epilogue:
//   sq    = max(nx + ny - 2*dot, 0)
//   denom = (1-nx)*(1-ny) + 1e-13
//   d     = max(1 + 2*sq/denom, 1+1e-6)
//   out   = -acosh(d) = -log(d + sqrt(d^2 - 1))
//
// One wave per 32(M) x 64(N) output block: 2 M-tiles x 4 N-tiles = 8
// accumulator tiles. Per K-step: 24 b128 loads vs 24 WMMAs (A frags reused
// 4x, B frags 2x) -> ~32 FLOP/byte from L2, keeping the WMMA pipe the
// bottleneck as the roofline demands.
// Grid: 256 m-strips * 64 n-strips = 16384 waves = 2048 blocks * 8 waves.
// ---------------------------------------------------------------------------
__global__ __launch_bounds__(256) void hyper_gemm_kernel(
    const unsigned short* __restrict__ xhi,
    const unsigned short* __restrict__ xlo,
    const unsigned short* __restrict__ yhi,
    const unsigned short* __restrict__ ylo,
    const float* __restrict__ nx,
    const float* __restrict__ ny,
    float* __restrict__ out)
{
    const int D = 256;   // K dimension
    const int L = 4096;  // output columns

    const int wave    = (blockIdx.x << 3) + (threadIdx.x >> 5);
    const int lane    = threadIdx.x & 31;
    const int half    = lane >> 4;   // 0: K-low chunk, 1: K-high chunk
    const int l15     = lane & 15;   // M row (A) / N col (B)
    const int m_strip = wave >> 6;   // 0..255 (32 rows each)
    const int n_strip = wave & 63;   // 0..63  (64 cols each)
    const int m_base  = m_strip * 32;
    const int n_base  = n_strip * 64;

    const unsigned short* xh0 = xhi + (m_base + l15) * D;
    const unsigned short* xl0 = xlo + (m_base + l15) * D;
    const unsigned short* xh1 = xh0 + 16 * D;
    const unsigned short* xl1 = xl0 + 16 * D;

    v8f acc[2][4];
    #pragma unroll
    for (int mt = 0; mt < 2; ++mt)
        #pragma unroll
        for (int t = 0; t < 4; ++t)
            acc[mt][t] = (v8f){0.f, 0.f, 0.f, 0.f, 0.f, 0.f, 0.f, 0.f};

    for (int kb = 0; kb < D; kb += 32) {
        const v16bf ah0 = load_frag(xh0, kb, half);
        const v16bf al0 = load_frag(xl0, kb, half);
        const v16bf ah1 = load_frag(xh1, kb, half);
        const v16bf al1 = load_frag(xl1, kb, half);
        #pragma unroll
        for (int t = 0; t < 4; ++t) {
            const unsigned short* yhr = yhi + (n_base + t * 16 + l15) * D;
            const unsigned short* ylr = ylo + (n_base + t * 16 + l15) * D;
            const v16bf bh = load_frag(yhr, kb, half);
            const v16bf bl = load_frag(ylr, kb, half);
            acc[0][t] = __builtin_amdgcn_wmma_f32_16x16x32_bf16(
                false, ah0, false, bh, (short)0, acc[0][t], false, false);
            acc[0][t] = __builtin_amdgcn_wmma_f32_16x16x32_bf16(
                false, ah0, false, bl, (short)0, acc[0][t], false, false);
            acc[0][t] = __builtin_amdgcn_wmma_f32_16x16x32_bf16(
                false, al0, false, bh, (short)0, acc[0][t], false, false);
            acc[1][t] = __builtin_amdgcn_wmma_f32_16x16x32_bf16(
                false, ah1, false, bh, (short)0, acc[1][t], false, false);
            acc[1][t] = __builtin_amdgcn_wmma_f32_16x16x32_bf16(
                false, ah1, false, bl, (short)0, acc[1][t], false, false);
            acc[1][t] = __builtin_amdgcn_wmma_f32_16x16x32_bf16(
                false, al1, false, bh, (short)0, acc[1][t], false, false);
        }
    }

    // Epilogue. C/D layout: lane (l15) -> N column; VGPR r -> M = r + half*8.
    #pragma unroll
    for (int mt = 0; mt < 2; ++mt) {
        float nxv[8], one_m_nx[8];
        const int m0 = m_base + mt * 16 + half * 8;
        #pragma unroll
        for (int r = 0; r < 8; ++r) {
            nxv[r]      = nx[m0 + r];
            one_m_nx[r] = 1.0f - nxv[r];
        }
        #pragma unroll
        for (int t = 0; t < 4; ++t) {
            const int   n        = n_base + t * 16 + l15;
            const float nyv      = ny[n];
            const float one_m_ny = 1.0f - nyv;
            #pragma unroll
            for (int r = 0; r < 8; ++r) {
                const int   m     = m0 + r;
                const float dot   = acc[mt][t][r];
                float sq          = nxv[r] + nyv - 2.0f * dot;
                sq                = fmaxf(sq, 0.0f);
                const float denom = one_m_nx[r] * one_m_ny + 1e-13f;
                float d           = 1.0f + 2.0f * sq / denom;
                d                 = fmaxf(d, 1.0f + 1e-6f);
                out[(size_t)m * L + n] = -logf(d + sqrtf(d * d - 1.0f));
            }
        }
    }
}

// ---------------------------------------------------------------------------
// Launch: d_in[0] = y (4096x256 f32), d_in[1] = x (8192x256 f32),
// d_out = (8192x4096) f32. Workspace layout (needs ~12.1 MB):
//   xhi 4MB | xlo 4MB | yhi 2MB | ylo 2MB | nx 32KB | ny 16KB
// ---------------------------------------------------------------------------
extern "C" void kernel_launch(void* const* d_in, const int* in_sizes, int n_in,
                              void* d_out, int out_size, void* d_ws, size_t ws_size,
                              hipStream_t stream) {
    (void)in_sizes; (void)n_in; (void)out_size; (void)ws_size;
    const float* y = (const float*)d_in[0];  // (4096, 256)
    const float* x = (const float*)d_in[1];  // (8192, 256)

    char* ws = (char*)d_ws;
    unsigned short* xhi = (unsigned short*)ws;
    unsigned short* xlo = xhi + (size_t)8192 * 256;
    unsigned short* yhi = xlo + (size_t)8192 * 256;
    unsigned short* ylo = yhi + (size_t)4096 * 256;
    float* nx = (float*)(ylo + (size_t)4096 * 256);
    float* ny = nx + 8192;

    proj_split_kernel<<<8192, 256, 0, stream>>>(x, xhi, xlo, nx);
    proj_split_kernel<<<4096, 256, 0, stream>>>(y, yhi, ylo, ny);
    hyper_gemm_kernel<<<2048, 256, 0, stream>>>(xhi, xlo, yhi, ylo, nx, ny,
                                                (float*)d_out);
}
